// LuongAttention_68891275428027
// MI455X (gfx1250) — matrix operand order repaired and checked by reading, not compiled, not run on hardware
//
#include <hip/hip_runtime.h>

typedef __attribute__((ext_vector_type(16))) _Float16 v16h;
typedef __attribute__((ext_vector_type(8)))  _Float16 v8h;
typedef __attribute__((ext_vector_type(8)))  float    v8f;

#define B_  4
#define QN  2048
#define KN  2048
#define DN  1024

// ---------------------------------------------------------------------------
// Kernel 0: fp32 -> f16 bulk convert (vectorized: 4 elems / thread)
// ---------------------------------------------------------------------------
__global__ void la_cvt_f16_kernel(const float* __restrict__ src,
                                  _Float16* __restrict__ dst, int n) {
  int base = (blockIdx.x * blockDim.x + threadIdx.x) * 4;
  if (base + 3 < n) {
    float4 f = *(const float4*)(src + base);
    dst[base + 0] = (_Float16)f.x;
    dst[base + 1] = (_Float16)f.y;
    dst[base + 2] = (_Float16)f.z;
    dst[base + 3] = (_Float16)f.w;
  }
}

// ---------------------------------------------------------------------------
// Kernel 1: V [b][k][d] fp32 -> Vt [b][d][k] f16, LDS-tiled 32x32 transpose
// ---------------------------------------------------------------------------
__global__ void la_transpose_kernel(const float* __restrict__ V,
                                    _Float16* __restrict__ Vt) {
  __shared__ _Float16 tile[32][33];           // +1 pad: avoid bank conflicts
  int b  = blockIdx.z;
  int k0 = blockIdx.x * 32;
  int d0 = blockIdx.y * 32;
  const float* Vb = V  + (size_t)b * KN * DN;
  _Float16*    Tb = Vt + (size_t)b * DN * KN;
  #pragma unroll
  for (int r = 0; r < 32; r += 8) {
    int k = k0 + threadIdx.y + r;
    int d = d0 + threadIdx.x;
    tile[threadIdx.y + r][threadIdx.x] = (_Float16)Vb[(size_t)k * DN + d];
  }
  __syncthreads();
  #pragma unroll
  for (int r = 0; r < 32; r += 8) {
    int d = d0 + threadIdx.y + r;
    int k = k0 + threadIdx.x;
    Tb[(size_t)d * KN + k] = tile[threadIdx.x][threadIdx.y + r];
  }
}

// ---------------------------------------------------------------------------
// Fragment loaders (16-bit operands, wave32): contiguous k order, same in A and B (not ISA table order).
//  A (16x32, MxK): lane m=L%16, half=L/16; half-wave covers
//                  K=half*16+0..15 -> one contiguous 32B chunk.
//  B (32x16, KxN): lane n=L%16; half-wave covers K=half*16+0..15 packed
//                  2-per-VGPR -> one contiguous 32B chunk.
// ---------------------------------------------------------------------------
__device__ __forceinline__ v16h load_a_frag(const _Float16* row, int aoff) {
  return *(const v16h*)(row + aoff);
}
__device__ __forceinline__ v16h load_b_frag(const _Float16* row, int boff) {
  return *(const v16h*)(row + boff);
}

// ---------------------------------------------------------------------------
// Kernel 2: scores S[b][q][k] = sum_d Qh[b][q][d] * Vh[b][k][d]   (f16 out)
// Block = 256 threads = 8 waves arranged 2(q) x 4(k); each wave computes a
// 32x64 output patch as a 2x4 grid of 16x16 WMMA tiles (8 accumulators),
// reusing A frags across 4 B frags: 12 b128 loads per 8 WMMAs.
// Block tile: 64q x 256k. Grid = B * (QN/64) * (KN/256) = 1024.
// ---------------------------------------------------------------------------
__global__ void la_scores_kernel(const _Float16* __restrict__ Qh,
                                 const _Float16* __restrict__ Vh,
                                 _Float16* __restrict__ S) {
  const int wave = threadIdx.x >> 5;
  const int lane = threadIdx.x & 31;
  const int kb = blockIdx.x % (KN / 256);
  const int qt = (blockIdx.x / (KN / 256)) % (QN / 64);
  const int b  = blockIdx.x / ((KN / 256) * (QN / 64));
  const int q0 = qt * 64 + (wave >> 2) * 32;     // wave's 32 q rows
  const int k0 = kb * 256 + (wave & 3) * 64;     // wave's 64 k cols

  const int m    = lane & 15;
  const int half = lane >> 4;
  const int aoff = half * 16;
  const int boff = half * 16;

  const _Float16* Qb = Qh + (size_t)b * QN * DN;
  const _Float16* Vb = Vh + (size_t)b * KN * DN;
  const _Float16* qrow[2];
  const _Float16* vrow[4];
  #pragma unroll
  for (int i = 0; i < 2; ++i) qrow[i] = Qb + (size_t)(q0 + i * 16 + m) * DN;
  #pragma unroll
  for (int j = 0; j < 4; ++j) vrow[j] = Vb + (size_t)(k0 + j * 16 + m) * DN;

  v8f c[2][4];
  #pragma unroll
  for (int i = 0; i < 2; ++i)
    #pragma unroll
    for (int j = 0; j < 4; ++j) c[i][j] = (v8f){};

  for (int d0 = 0; d0 < DN; d0 += 32) {
    v16h a[2], bb[4];
    #pragma unroll
    for (int i = 0; i < 2; ++i) a[i]  = load_a_frag(qrow[i] + d0, aoff);
    #pragma unroll
    for (int j = 0; j < 4; ++j) bb[j] = load_b_frag(vrow[j] + d0, boff);
    #pragma unroll
    for (int i = 0; i < 2; ++i)
      #pragma unroll
      for (int j = 0; j < 4; ++j)
        c[i][j] = __builtin_amdgcn_wmma_f32_16x16x32_f16(
            false, a[i], false, bb[j], (short)0, c[i][j], false, false);
  }

  _Float16* Sb = S + (size_t)b * QN * KN;
  #pragma unroll
  for (int i = 0; i < 2; ++i)
    #pragma unroll
    for (int j = 0; j < 4; ++j)
      #pragma unroll
      for (int r = 0; r < 8; ++r) {
        int row = q0 + i * 16 + r + 8 * half;    // C layout: VGPR r -> M=r / r+8
        Sb[(size_t)row * KN + k0 + j * 16 + m] = (_Float16)c[i][j][r];
      }
}

// ---------------------------------------------------------------------------
// Kernel 3: in-place row softmax over K. One block (256 thr) per (b,q) row.
// ---------------------------------------------------------------------------
__global__ void la_softmax_kernel(_Float16* __restrict__ S) {
  __shared__ float red[256];
  const int t = threadIdx.x;
  _Float16* p = S + (size_t)blockIdx.x * KN;

  float v[8];
  float mx = -3.0e38f;
  #pragma unroll
  for (int j = 0; j < 8; ++j) {
    v[j] = (float)p[t * 8 + j];
    mx = fmaxf(mx, v[j]);
  }
  red[t] = mx; __syncthreads();
  for (int s = 128; s > 0; s >>= 1) {
    if (t < s) red[t] = fmaxf(red[t], red[t + s]);
    __syncthreads();
  }
  mx = red[0]; __syncthreads();

  float sum = 0.0f;
  #pragma unroll
  for (int j = 0; j < 8; ++j) { v[j] = __expf(v[j] - mx); sum += v[j]; }
  red[t] = sum; __syncthreads();
  for (int s = 128; s > 0; s >>= 1) {
    if (t < s) red[t] += red[t + s];
    __syncthreads();
  }
  float inv = 1.0f / red[0];
  #pragma unroll
  for (int j = 0; j < 8; ++j) p[t * 8 + j] = (_Float16)(v[j] * inv);
}

// ---------------------------------------------------------------------------
// Kernel 4: context O[b][q][d] = sum_k P[b][q][k] * V[b][k][d]
// A = P rows (f16), B rows from Vt[b][d][k] (contiguous along k).
// Same 2x4 register blocking: wave computes 32q x 64d, 8 WMMAs / k-step.
// Block tile: 64q x 256d. Grid = B * (QN/64) * (DN/256) = 512.
// ---------------------------------------------------------------------------
__global__ void la_context_kernel(const _Float16* __restrict__ P,
                                  const _Float16* __restrict__ Vt,
                                  float* __restrict__ O) {
  const int wave = threadIdx.x >> 5;
  const int lane = threadIdx.x & 31;
  const int dt = blockIdx.x % (DN / 256);
  const int qt = (blockIdx.x / (DN / 256)) % (QN / 64);
  const int b  = blockIdx.x / ((DN / 256) * (QN / 64));
  const int q0 = qt * 64 + (wave >> 2) * 32;     // wave's 32 q rows
  const int d0 = dt * 256 + (wave & 3) * 64;     // wave's 64 d cols

  const int m    = lane & 15;
  const int half = lane >> 4;
  const int aoff = half * 16;
  const int boff = half * 16;

  const _Float16* Pb = P  + (size_t)b * QN * KN;
  const _Float16* Tb = Vt + (size_t)b * DN * KN;
  const _Float16* prow[2];
  const _Float16* vtrow[4];
  #pragma unroll
  for (int i = 0; i < 2; ++i) prow[i]  = Pb + (size_t)(q0 + i * 16 + m) * KN;
  #pragma unroll
  for (int j = 0; j < 4; ++j) vtrow[j] = Tb + (size_t)(d0 + j * 16 + m) * KN;

  v8f c[2][4];
  #pragma unroll
  for (int i = 0; i < 2; ++i)
    #pragma unroll
    for (int j = 0; j < 4; ++j) c[i][j] = (v8f){};

  for (int k0 = 0; k0 < KN; k0 += 32) {
    v16h a[2], bb[4];
    #pragma unroll
    for (int i = 0; i < 2; ++i) a[i]  = load_a_frag(prow[i] + k0, aoff);
    #pragma unroll
    for (int j = 0; j < 4; ++j) bb[j] = load_b_frag(vtrow[j] + k0, boff);
    #pragma unroll
    for (int i = 0; i < 2; ++i)
      #pragma unroll
      for (int j = 0; j < 4; ++j)
        c[i][j] = __builtin_amdgcn_wmma_f32_16x16x32_f16(
            false, a[i], false, bb[j], (short)0, c[i][j], false, false);
  }

  float* Ob = O + (size_t)b * QN * DN;
  #pragma unroll
  for (int i = 0; i < 2; ++i)
    #pragma unroll
    for (int j = 0; j < 4; ++j)
      #pragma unroll
      for (int r = 0; r < 8; ++r)
        Ob[(size_t)(q0 + i * 16 + r + 8 * half) * DN + d0 + j * 16 + m] =
            c[i][j][r];
}

// ---------------------------------------------------------------------------
extern "C" void kernel_launch(void* const* d_in, const int* in_sizes, int n_in,
                              void* d_out, int out_size, void* d_ws, size_t ws_size,
                              hipStream_t stream) {
  const float* Qf = (const float*)d_in[0];   // [B,Q,D] fp32
  const float* Vf = (const float*)d_in[1];   // [B,K,D] fp32
  float* Out = (float*)d_out;                // [B,Q,D] fp32

  char* ws = (char*)d_ws;
  _Float16* S  = (_Float16*)(ws);                                   // 32 MB: scores -> probs
  _Float16* Qh = (_Float16*)(ws + (size_t)32 * 1024 * 1024);        // 16 MB
  _Float16* Vh = (_Float16*)(ws + (size_t)48 * 1024 * 1024);        // 16 MB
  _Float16* Vt = (_Float16*)(ws + (size_t)64 * 1024 * 1024);        // 16 MB

  const int nElem = B_ * QN * DN;  // == B_*KN*DN == 8M

  la_cvt_f16_kernel<<<nElem / 4 / 256, 256, 0, stream>>>(Qf, Qh, nElem);
  la_cvt_f16_kernel<<<nElem / 4 / 256, 256, 0, stream>>>(Vf, Vh, nElem);

  dim3 tgrid(KN / 32, DN / 32, B_);
  la_transpose_kernel<<<tgrid, dim3(32, 8), 0, stream>>>(Vf, Vt);

  la_scores_kernel<<<B_ * (QN / 64) * (KN / 256), 256, 0, stream>>>(Qh, Vh, S);
  la_softmax_kernel<<<B_ * QN, 256, 0, stream>>>(S);
  la_context_kernel<<<B_ * (QN / 64) * (DN / 256), 256, 0, stream>>>(S, Vt, Out);
}
